// MultiHeadAttention_20856361189814
// MI455X (gfx1250) — matrix operand order repaired and checked by reading, not compiled, not run on hardware
//
#include <hip/hip_runtime.h>
#include <hip/hip_bf16.h>

// ---------------------------------------------------------------------------
// MultiHeadAttention forward for MI455X (gfx1250, wave32, WMMA bf16)
// B=4, T=2048, D=1024, H=16, hd=64
// ---------------------------------------------------------------------------

#define B_SZ 4
#define T_SZ 2048
#define D_SZ 1024
#define H_SZ 16
#define HD_SZ 64
#define SCALE 0.125f // 64^-0.5

typedef __attribute__((ext_vector_type(16))) __bf16 v16bf;
typedef __attribute__((ext_vector_type(8)))  float  v8f;
typedef __attribute__((ext_vector_type(4)))  float  vf4;

union FragU { v16bf v; vf4 f[2]; };

static __device__ inline v8f vzero8() {
  v8f z = {0.f,0.f,0.f,0.f,0.f,0.f,0.f,0.f};
  return z;
}

static __device__ inline __bf16 f2bf(float f) {
  union { float f; unsigned u; } uf; uf.f = f;
  unsigned u = uf.u;
  unsigned r = u + 0x7FFFu + ((u >> 16) & 1u); // round-to-nearest-even
  union { unsigned short s; __bf16 b; } ub; ub.s = (unsigned short)(r >> 16);
  return ub.b;
}

// A-operand fragment (16x32, M x K), row-major source, ld = row stride.
// Layout per ISA 7.12.2: lane L (L<16) holds row L, K = {0..7, 16..23};
// lane L+16 holds row L, K = {8..15, 24..31}.
static __device__ inline v16bf load_frag_a(const __bf16* base, int row, int ld,
                                           int k0, int lane) {
  FragU u;
  const __bf16* p = base + (size_t)row * ld + k0 + ((lane >> 4) << 3);
  u.f[0] = *(const vf4*)(p);
  u.f[1] = *(const vf4*)(p + 16);
  return u.v;
}

// B-operand fragment (32x16, K x N) from a row-major [N, K] ("Bt") source.
// lanes 0-15: n = lane, K = {0..7, 16..23} ; lanes 16-31: n = lane-16, K = {8..15, 24..31}.
static __device__ inline v16bf load_frag_b(const __bf16* baseT, int n0, int ld,
                                           int k0, int lane) {
  FragU u;
  const __bf16* p = baseT + (size_t)(n0 + (lane & 15)) * ld + k0 + ((lane >> 4) << 3);
  u.f[0] = *(const vf4*)(p);
  u.f[1] = *(const vf4*)(p + 16);
  return u.v;
}

static __device__ inline v8f wmma_bf16(v16bf a, v16bf b, v8f c) {
  return __builtin_amdgcn_wmma_f32_16x16x32_bf16(
      /*neg_a=*/false, a, /*neg_b=*/false, b,
      /*c_mod=*/(short)0, c, /*reuse_a=*/false, /*reuse_b=*/false);
}

// Async global -> LDS copy of 16 bytes per lane (GV addressing, ASYNCcnt).
static __device__ inline void async_copy16(const void* gptr, void* lptr) {
  unsigned lds = (unsigned)(unsigned long long)lptr; // low 32b = LDS offset
  asm volatile("global_load_async_to_lds_b128 %0, %1, off"
               :: "v"(lds), "v"(gptr) : "memory");
}

static __device__ inline float redmax16(float v) {
#pragma unroll
  for (int m = 1; m < 16; m <<= 1) v = fmaxf(v, __shfl_xor(v, m, 32));
  return v;
}
static __device__ inline float redsum16(float v) {
#pragma unroll
  for (int m = 1; m < 16; m <<= 1) v += __shfl_xor(v, m, 32);
  return v;
}

// ---------------------------------------------------------------------------
// fp32 -> bf16 conversion
// ---------------------------------------------------------------------------
__global__ void cvt_bf16_kernel(const float* __restrict__ in,
                                __bf16* __restrict__ out, int n) {
  int i = blockIdx.x * blockDim.x + threadIdx.x;
  int stride = gridDim.x * blockDim.x;
  for (; i < n; i += stride) out[i] = f2bf(in[i]);
}

// ---------------------------------------------------------------------------
// QKV GEMM: A = x_bf16 [8192, 1024], Bt = Wqkv_bf16 [3072, 1024]
// Epilogue scatters into Q [B,H,T,64], K [B,H,T,64], Vt [B,H,64,T] (bf16).
// Block: 256 thr (8 waves), tile 128(M) x 128(N); wave tile 32 x 64.
// ---------------------------------------------------------------------------
__global__ void qkv_gemm_kernel(const __bf16* __restrict__ A,
                                const __bf16* __restrict__ Bt,
                                __bf16* __restrict__ Qb,
                                __bf16* __restrict__ Kb,
                                __bf16* __restrict__ Vtb) {
  const int K = D_SZ;
  const int lane = threadIdx.x & 31;
  const int wave = threadIdx.x >> 5;
  const int m0 = blockIdx.x * 128 + (wave >> 1) * 32;
  const int n0 = blockIdx.y * 128 + (wave & 1) * 64;

  v8f acc[2][4];
#pragma unroll
  for (int i = 0; i < 2; ++i)
#pragma unroll
    for (int j = 0; j < 4; ++j) acc[i][j] = vzero8();

  const int arow0 = m0 + (lane & 15);
  for (int k0 = 0; k0 < K; k0 += 32) {
    __builtin_prefetch(A + (size_t)arow0 * K + k0 + 64, 0, 1);
    v16bf a0 = load_frag_a(A, arow0, K, k0, lane);
    v16bf a1 = load_frag_a(A, arow0 + 16, K, k0, lane);
#pragma unroll
    for (int j = 0; j < 4; ++j) {
      v16bf b = load_frag_b(Bt, n0 + 16 * j, K, k0, lane);
      acc[0][j] = wmma_bf16(a0, b, acc[0][j]);
      acc[1][j] = wmma_bf16(a1, b, acc[1][j]);
    }
  }

  // Epilogue: C element (M = m0+i*16 + r + 8*(lane>>4), N = n0+j*16 + lane&15)
  const int hi = lane >> 4;
#pragma unroll
  for (int i = 0; i < 2; ++i) {
#pragma unroll
    for (int j = 0; j < 4; ++j) {
      const int n = n0 + 16 * j + (lane & 15);
      const int which = n >> 10;            // 0=Q 1=K 2=V
      const int e = n & 1023;
      const int h = e >> 6;
      const int d = e & 63;
#pragma unroll
      for (int r = 0; r < 8; ++r) {
        const int m = m0 + 16 * i + r + 8 * hi;
        const int b = m >> 11;
        const int t = m & 2047;
        const __bf16 bv = f2bf(acc[i][j][r]);
        const size_t bh = (size_t)(b * H_SZ + h);
        if (which == 0)      Qb [(bh * T_SZ + t) * HD_SZ + d] = bv;
        else if (which == 1) Kb [(bh * T_SZ + t) * HD_SZ + d] = bv;
        else                 Vtb[(bh * HD_SZ + d) * T_SZ + t] = bv;
      }
    }
  }
}

// ---------------------------------------------------------------------------
// Flash attention: grid (B*H, T/64), block 128 (4 waves, 16 q-rows each).
// K/V tiles staged into LDS once per block via async copies and shared by all
// 4 waves; S = Q K^T * SCALE (causal), online softmax, O += P V.
// P (16x32 fp32 C-layout) is transposed to the A-operand layout through LDS.
// ---------------------------------------------------------------------------
__global__ void attn_kernel(const __bf16* __restrict__ Qb,
                            const __bf16* __restrict__ Kb,
                            const __bf16* __restrict__ Vtb,
                            __bf16* __restrict__ Ob) {
  const int bh = blockIdx.x;                 // b*16 + h
  const int q0 = blockIdx.y * 64;
  const int tid = threadIdx.x;
  const int wave = tid >> 5;
  const int lane = tid & 31;
  const int hi = lane >> 4;
  const int qrow0 = q0 + wave * 16;

  const __bf16* Qp = Qb  + (size_t)bh * T_SZ * HD_SZ;  // [T, 64]
  const __bf16* Kp = Kb  + (size_t)bh * T_SZ * HD_SZ;  // [T, 64]
  const __bf16* Vp = Vtb + (size_t)bh * HD_SZ * T_SZ;  // [64, T]

  __shared__ __align__(16) __bf16 Kt[32][64];          // keys tile [key][d]
  __shared__ __align__(16) __bf16 Vt[64][32];          // values tile [d][key]
  __shared__ __align__(16) __bf16 Plds[4][16][32];     // per-wave P staging

  // Q fragments: 16 rows x 64 K  ->  two 16x32 A fragments, loaded once.
  v16bf qa[2];
#pragma unroll
  for (int c = 0; c < 2; ++c)
    qa[c] = load_frag_a(Qp, qrow0 + (lane & 15), HD_SZ, c * 32, lane);

  v8f o[4];
#pragma unroll
  for (int nt = 0; nt < 4; ++nt) o[nt] = vzero8();
  float m_i[8], l_i[8];
#pragma unroll
  for (int r = 0; r < 8; ++r) { m_i[r] = -1e30f; l_i[r] = 0.f; }

  // Block-uniform causal bound so the LDS barriers are non-divergent; waves
  // with tighter bounds see fully-masked tiles (p == 0 contributions).
  const int jmax = q0 + 63;
  for (int j = 0; j <= jmax; j += 32) {
    __syncthreads();  // previous iteration's LDS reads complete

    // ---- cooperative async staging of K tile (rows j..j+31 contiguous,
    //      4 KB) and V tile (64 rows x 32 keys from Vt, 4 KB)
    {
      const char* gK = (const char*)(Kp + (size_t)j * HD_SZ);
      char* lK = (char*)&Kt[0][0];
      async_copy16(gK + tid * 32,      lK + tid * 32);
      async_copy16(gK + tid * 32 + 16, lK + tid * 32 + 16);

      const int dv = tid >> 1;          // 0..63
      const int half = tid & 1;         // 16-element half of the 32-key row
      const char* gV = (const char*)(Vp + (size_t)dv * T_SZ + j + half * 16);
      char* lV = (char*)&Vt[dv][half * 16];
      async_copy16(gV,      lV);
      async_copy16(gV + 16, lV + 16);
    }
    asm volatile("s_wait_asynccnt 0x0" ::: "memory");
    __syncthreads();  // tiles visible to all waves

    // ---- S = Q K^T : two 16x16 tiles (keys j..j+15, j+16..j+31), from LDS
    v8f s0 = vzero8(), s1 = vzero8();
#pragma unroll
    for (int c = 0; c < 2; ++c) {
      v16bf kb0 = load_frag_b(&Kt[0][0], 0,  HD_SZ, c * 32, lane);
      v16bf kb1 = load_frag_b(&Kt[0][0], 16, HD_SZ, c * 32, lane);
      s0 = wmma_bf16(qa[c], kb0, s0);
      s1 = wmma_bf16(qa[c], kb1, s1);
    }

    // ---- scale + causal mask + online softmax update
    const int col0 = j + (lane & 15);
    const int col1 = j + 16 + (lane & 15);
#pragma unroll
    for (int r = 0; r < 8; ++r) {
      const int qr = qrow0 + r + 8 * hi;
      float v0 = s0[r] * SCALE; if (col0 > qr) v0 = -1e30f;
      float v1 = s1[r] * SCALE; if (col1 > qr) v1 = -1e30f;
      const float rowmax = redmax16(fmaxf(v0, v1));
      const float mnew = fmaxf(m_i[r], rowmax);
      const float corr = __expf(m_i[r] - mnew);
      const float p0 = __expf(v0 - mnew);
      const float p1 = __expf(v1 - mnew);
      l_i[r] = l_i[r] * corr + redsum16(p0 + p1);
      m_i[r] = mnew;
#pragma unroll
      for (int nt = 0; nt < 4; ++nt) o[nt][r] *= corr;
      const int prow = r + 8 * hi;
      Plds[wave][prow][lane & 15]        = f2bf(p0);
      Plds[wave][prow][16 + (lane & 15)] = f2bf(p1);
    }

    // LDS writes are wave-private; wait on DS counter before re-reading.
    asm volatile("s_wait_dscnt 0" ::: "memory");

    // ---- re-read P in A-operand layout (16x32)
    FragU pf;
    const int prow = lane & 15;
    pf.f[0] = *(const vf4*)&Plds[wave][prow][hi * 8];
    pf.f[1] = *(const vf4*)&Plds[wave][prow][16 + hi * 8];

    // ---- O += P V : V tile is [d][key] in LDS, contiguous B fragments
#pragma unroll
    for (int nt = 0; nt < 4; ++nt) {
      v16bf vb = load_frag_b(&Vt[0][0], nt * 16, 32, 0, lane);
      o[nt] = wmma_bf16(pf.v, vb, o[nt]);
    }
  }

  // ---- epilogue: O /= l, write bf16 into attn buffer [B, T, 1024]
  const int b = bh >> 4;
  const int h = bh & 15;
#pragma unroll
  for (int nt = 0; nt < 4; ++nt) {
#pragma unroll
    for (int r = 0; r < 8; ++r) {
      const int q = qrow0 + r + 8 * hi;
      const int d = nt * 16 + (lane & 15);
      const float val = o[nt][r] / l_i[r];
      Ob[((size_t)(b * T_SZ + q)) * D_SZ + h * HD_SZ + d] = f2bf(val);
    }
  }
}

// ---------------------------------------------------------------------------
// Output projection: A = attn_bf16 [8192, 1024], Bt = Wproj_bf16 [1024, 1024]
// fp32 output to d_out [8192, 1024].
// ---------------------------------------------------------------------------
__global__ void proj_gemm_kernel(const __bf16* __restrict__ A,
                                 const __bf16* __restrict__ Bt,
                                 float* __restrict__ out) {
  const int K = D_SZ;
  const int N = D_SZ;
  const int lane = threadIdx.x & 31;
  const int wave = threadIdx.x >> 5;
  const int m0 = blockIdx.x * 128 + (wave >> 1) * 32;
  const int n0 = blockIdx.y * 128 + (wave & 1) * 64;

  v8f acc[2][4];
#pragma unroll
  for (int i = 0; i < 2; ++i)
#pragma unroll
    for (int j = 0; j < 4; ++j) acc[i][j] = vzero8();

  const int arow0 = m0 + (lane & 15);
  for (int k0 = 0; k0 < K; k0 += 32) {
    __builtin_prefetch(A + (size_t)arow0 * K + k0 + 64, 0, 1);
    v16bf a0 = load_frag_a(A, arow0, K, k0, lane);
    v16bf a1 = load_frag_a(A, arow0 + 16, K, k0, lane);
#pragma unroll
    for (int j = 0; j < 4; ++j) {
      v16bf b = load_frag_b(Bt, n0 + 16 * j, K, k0, lane);
      acc[0][j] = wmma_bf16(a0, b, acc[0][j]);
      acc[1][j] = wmma_bf16(a1, b, acc[1][j]);
    }
  }

  const int hi = lane >> 4;
#pragma unroll
  for (int i = 0; i < 2; ++i)
#pragma unroll
    for (int j = 0; j < 4; ++j) {
      const int n = n0 + 16 * j + (lane & 15);
#pragma unroll
      for (int r = 0; r < 8; ++r) {
        const int m = m0 + 16 * i + r + 8 * hi;
        out[(size_t)m * N + n] = acc[i][j][r];
      }
    }
}

// ---------------------------------------------------------------------------
// Launch
// ---------------------------------------------------------------------------
extern "C" void kernel_launch(void* const* d_in, const int* in_sizes, int n_in,
                              void* d_out, int out_size, void* d_ws, size_t ws_size,
                              hipStream_t stream) {
  const float* x     = (const float*)d_in[0];  // [4, 2048, 1024]
  const float* Wqkv  = (const float*)d_in[1];  // [3072, 1024]
  const float* Wproj = (const float*)d_in[2];  // [1024, 1024]
  float* out = (float*)d_out;                  // [4, 2048, 1024]

  const size_t n_x    = (size_t)B_SZ * T_SZ * D_SZ;      // 8388608
  const size_t n_wqkv = (size_t)3 * D_SZ * D_SZ;         // 3145728
  const size_t n_wprj = (size_t)D_SZ * D_SZ;             // 1048576
  const size_t n_qkv  = (size_t)B_SZ * H_SZ * T_SZ * HD_SZ; // 8388608

  __bf16* ws = (__bf16*)d_ws;
  size_t off = 0;
  __bf16* xb     = ws + off; off += n_x;
  __bf16* wqkvb  = ws + off; off += n_wqkv;
  __bf16* wprojb = ws + off; off += n_wprj;
  __bf16* Qb     = ws + off; off += n_qkv;
  __bf16* Kb     = ws + off; off += n_qkv;
  __bf16* Vtb    = ws + off; off += n_qkv;
  __bf16* Ab     = ws + off; off += n_x;

  cvt_bf16_kernel<<<1024, 256, 0, stream>>>(x, xb, (int)n_x);
  cvt_bf16_kernel<<<1024, 256, 0, stream>>>(Wqkv, wqkvb, (int)n_wqkv);
  cvt_bf16_kernel<<<512, 256, 0, stream>>>(Wproj, wprojb, (int)n_wprj);

  // QKV GEMM: M=8192, N=3072 -> grid (64, 24)
  qkv_gemm_kernel<<<dim3(64, 24), 256, 0, stream>>>(xb, wqkvb, Qb, Kb, Vtb);

  // Attention: grid (B*H=64, T/64=32), 128 threads
  attn_kernel<<<dim3(64, 32), 128, 0, stream>>>(Qb, Kb, Vtb, Ab);

  // Output projection: M=8192, N=1024 -> grid (64, 8)
  proj_gemm_kernel<<<dim3(64, 8), 256, 0, stream>>>(Ab, wprojb, out);
}
